// BTT_56788057588464
// MI455X (gfx1250) — compile-verified, matrix operand order checked by reading, not run on hardware
//
#include <hip/hip_runtime.h>
#include <hip/hip_bf16.h>

// ---------------------------------------------------------------------------
// BTT (2-core tensor-train) fused forward for MI455X (gfx1250, wave32, WMMA)
//   out[B, y*64+i] = bias + sum_{j,a} ( sum_x X[B, j*64+x] * c0[j,x,i,a] ) * c1[j,y,i,a]
// B=8192, D=64, R=4. bf16 WMMA 16x16x32, f32 accumulate.
// Block: 256 threads (8 waves). Tile: 16 batch rows x 8 i-values.
// LDS (128KB dynamic): 2 x 32KB X double-buffer (async-to-LDS) + 64KB bf16 z1.
// Output transposed through LDS and streamed with store-async-from-LDS.
// ---------------------------------------------------------------------------

typedef __attribute__((ext_vector_type(16))) __bf16 v16bf;
typedef __attribute__((ext_vector_type(8)))  float  v8f;
typedef __attribute__((ext_vector_type(4)))  int    v4i;

#define AS1 __attribute__((address_space(1)))
#define AS3 __attribute__((address_space(3)))

#if __has_builtin(__builtin_amdgcn_global_load_async_to_lds_b128)
#  define BTT_ASYNC_LD 1
#endif
#if __has_builtin(__builtin_amdgcn_global_store_async_from_lds_b128)
#  define BTT_ASYNC_ST 1
#endif

#if __has_builtin(__builtin_amdgcn_s_wait_asynccnt)
#  define BTT_WAIT_ASYNC0() __builtin_amdgcn_s_wait_asynccnt(0)
#else
#  define BTT_WAIT_ASYNC0() asm volatile("s_wait_asynccnt 0x0" ::: "memory")
#endif

#define BTT_B    8192
#define BTT_D    64
#define BTT_R    4
#define BTT_IT   8      // i-values per block
#define BTT_BT   16     // batch rows per block

// 16-bit A/B fragment K index for element h (0..15) of v16bf, hi = lane>=16
__device__ __forceinline__ int kmap(int h, int hi) {
    return (h & 7) + ((h & 8) << 1) + (hi << 3);
}

// Issue one 16x512 f32 X chunk into LDS (async-to-LDS when available).
__device__ __forceinline__ void btt_stage_issue(const float* __restrict__ x,
                                                float* __restrict__ dst,
                                                int b0, int jc, int tid) {
#pragma unroll
    for (int t = 0; t < 8; ++t) {
        int f   = t * 256 + tid;                    // float4 index, 2048 total
        int row = f >> 7;                           // 128 float4 per row
        int c4  = f & 127;
        const float* src = x + (size_t)(b0 + row) * 4096 + (size_t)jc * 512 + c4 * 4;
        float*       d   = dst + row * 512 + c4 * 4;
#ifdef BTT_ASYNC_LD
        __builtin_amdgcn_global_load_async_to_lds_b128((AS1 v4i*)src, (AS3 v4i*)d, 0, 0);
#else
        *(float4*)d = *(const float4*)src;
#endif
    }
}

__global__ __launch_bounds__(256)
void btt_fused_kernel(const float* __restrict__ x,
                      const float* __restrict__ core0,   // [j][x][y][a]  (D,D,D,R)
                      const float* __restrict__ core1,   // [j][y][i][a]  (D,D,D,R)
                      const float* __restrict__ bias,    // [4096]
                      float* __restrict__ out) {
    extern __shared__ char smem[];
    float*  xb0    = (float*)smem;                  // 16 x 512 f32 = 32KB (buffer 0)
    float*  xb1    = (float*)(smem + 32 * 1024);    // 16 x 512 f32 = 32KB (buffer 1)
    float*  lds_o  = (float*)smem;                  // phase-2 out staging (aliases xb0)
    __bf16* lds_z1 = (__bf16*)(smem + 64 * 1024);   // [il=8][m=16][k=256] bf16 = 64KB

    const int b0    = blockIdx.x * BTT_BT;          // batch row base
    const int iBase = blockIdx.y * BTT_IT;          // i base
    const int tid   = threadIdx.x;
    const int wave  = tid >> 5;                     // 0..7 (wave32)
    const int lane  = tid & 31;
    const int m16   = lane & 15;
    const int hi    = lane >> 4;

    // ---------------- Phase 1: z1[j, i, a] = X_j x C0_j, store bf16 to LDS --
    btt_stage_issue(x, xb0, b0, 0, tid);            // prologue: chunk 0
#ifdef BTT_ASYNC_LD
    BTT_WAIT_ASYNC0();
#endif
    __syncthreads();

    for (int jc = 0; jc < 8; ++jc) {
        float* cur = (jc & 1) ? xb1 : xb0;
        if (jc < 7)                                 // issue next chunk, overlap compute
            btt_stage_issue(x, (jc & 1) ? xb0 : xb1, b0, jc + 1, tid);

        const int j  = jc * 8 + wave;               // this wave's j
        // A fragments: X[m, j*64 + k], K = 0..63 split into two K=32 fragments
        v16bf A0, A1;
        #pragma unroll
        for (int h = 0; h < 16; ++h) {
            int k = kmap(h, hi);
            A0[h] = (__bf16)cur[m16 * 512 + wave * 64 + k];
            A1[h] = (__bf16)cur[m16 * 512 + wave * 64 + 32 + k];
        }

        const float* c0j = core0 + (size_t)j * (64 * 64 * 4);   // [x][y][a]
        #pragma unroll
        for (int nt = 0; nt < 2; ++nt) {            // N = 32 cols (8i x 4a) -> 2 tiles
            int n  = nt * 16 + m16;                 // col within (i_local, a) grid
            int il = n >> 2;                        // i_local 0..7
            int a  = n & 3;
            int y  = iBase + il;
            v16bf B0, B1;
            #pragma unroll
            for (int h = 0; h < 16; ++h) {
                int xk = kmap(h, hi);
                B0[h] = (__bf16)c0j[(size_t)xk * 256 + y * 4 + a];
                B1[h] = (__bf16)c0j[(size_t)(xk + 32) * 256 + y * 4 + a];
            }
            v8f acc = {};
            acc = __builtin_amdgcn_wmma_f32_16x16x32_bf16(false, A0, false, B0,
                                                          (short)0, acc, false, false);
            acc = __builtin_amdgcn_wmma_f32_16x16x32_bf16(false, A1, false, B1,
                                                          (short)0, acc, false, false);
            // scatter D tile into phase-2 A-fragment layout: [il][m][k = j*4+a]
            #pragma unroll
            for (int r = 0; r < 8; ++r) {
                int mm = r + (hi << 3);
                lds_z1[((il * 16 + mm) << 8) + (j << 2) + a] = (__bf16)acc[r];
            }
        }

        if (jc < 7) {
#ifdef BTT_ASYNC_LD
            BTT_WAIT_ASYNC0();                      // next chunk landed in LDS
#endif
            __syncthreads();
        }
    }
    __syncthreads();

    // ---------------- Phase 2: out[:, y, i] = z1[:, (j,a), i] x C1 ---------
    #pragma unroll
    for (int q = 0; q < 4; ++q) {
        int id = wave * 4 + q;                      // 32 subtiles: 8 i x 4 y-tiles
        int il = id >> 2;
        int yt = id & 3;
        int iG = iBase + il;
        int y16 = yt * 16 + m16;
        v8f accA = {}, accB = {};                   // two chains hide WMMA RAW hazard
        #pragma unroll
        for (int kc = 0; kc < 8; kc += 2) {
            v16bf Af0, Bf0, Af1, Bf1;
            #pragma unroll
            for (int h = 0; h < 16; ++h) {
                int k0 = kc * 32 + kmap(h, hi);     // k = j*4 + a
                int k1 = k0 + 32;
                Af0[h] = lds_z1[((il * 16 + m16) << 8) + k0];
                Af1[h] = lds_z1[((il * 16 + m16) << 8) + k1];
                Bf0[h] = (__bf16)core1[(size_t)(k0 >> 2) * 16384 + y16 * 256 + iG * 4 + (k0 & 3)];
                Bf1[h] = (__bf16)core1[(size_t)(k1 >> 2) * 16384 + y16 * 256 + iG * 4 + (k1 & 3)];
            }
            accA = __builtin_amdgcn_wmma_f32_16x16x32_bf16(false, Af0, false, Bf0,
                                                           (short)0, accA, false, false);
            accB = __builtin_amdgcn_wmma_f32_16x16x32_bf16(false, Af1, false, Bf1,
                                                           (short)0, accB, false, false);
        }
        // bias depends only on the output column (y, i)
        float bv = bias[y16 * 64 + iG];
        #pragma unroll
        for (int r = 0; r < 8; ++r) {
            int mm = r + (hi << 3);
            lds_o[mm * 512 + y16 * 8 + il] = accA[r] + accB[r] + bv;  // local col = y*8 + il
        }
    }
    __syncthreads();

    // cooperative store: 16 rows x 512 local cols -> out columns (stream from LDS)
    #pragma unroll
    for (int t = 0; t < 8; ++t) {
        int f    = t * 256 + tid;                   // float4 index, 2048 total
        int row  = f >> 7;
        int cl   = (f & 127) * 4;                   // local col, step 4
        int y    = cl >> 3;
        int ilq  = cl & 7;                          // 0 or 4 (stays in same y group)
        float* srcl = lds_o + row * 512 + cl;
        float* dstg = out + (size_t)(b0 + row) * 4096 + y * 64 + iBase + ilq;
#ifdef BTT_ASYNC_ST
        __builtin_amdgcn_global_store_async_from_lds_b128((AS1 v4i*)dstg, (AS3 v4i*)srcl, 0, 0);
#else
        *(float4*)dstg = *(float4*)srcl;
#endif
    }
    // S_ENDPGM performs an implicit wait-idle, covering outstanding async stores.
}

extern "C" void kernel_launch(void* const* d_in, const int* in_sizes, int n_in,
                              void* d_out, int out_size, void* d_ws, size_t ws_size,
                              hipStream_t stream) {
    (void)in_sizes; (void)n_in; (void)d_ws; (void)ws_size; (void)out_size;
    const float* x     = (const float*)d_in[0];
    const float* core0 = (const float*)d_in[1];
    const float* core1 = (const float*)d_in[2];
    const float* bias  = (const float*)d_in[3];
    float* out = (float*)d_out;

    dim3 grid(BTT_B / BTT_BT, BTT_D / BTT_IT);      // 512 x 8
    dim3 block(256);
    size_t shmem = 128 * 1024;                      // 2x32KB X + 64KB z1
    btt_fused_kernel<<<grid, block, shmem, stream>>>(x, core0, core1, bias, out);
}
